// NetSparse1_78718160601156
// MI455X (gfx1250) — matrix operand order repaired and checked by reading, not compiled, not run on hardware
//
#include <hip/hip_runtime.h>
#include <hip/hip_bf16.h>

// ---------------------------------------------------------------------------
// Sparse-MLP forward for gfx1250 (MI455X), bf16 WMMA + fused second GEMM.
// Round 3: fix async-to-LDS builtin pointer types (AS1/AS3 v4i*).
// ---------------------------------------------------------------------------

typedef __bf16 bf16t;
typedef bf16t v16bf __attribute__((ext_vector_type(16)));
typedef float v8f   __attribute__((ext_vector_type(8)));
typedef int   v4i   __attribute__((ext_vector_type(4)));

union FragU { uint4 q[2]; v16bf v; };

#define HIDDEN 8192
#define IN_DIM 784
#define KPAD   800          // 25 * 32
#define KT_N   25
#define OUT_DIM 10
#define BATCH  16384

#define AS1 __attribute__((address_space(1)))
#define AS3 __attribute__((address_space(3)))

__device__ __forceinline__ unsigned short f2bf(float f) {
    unsigned u = __float_as_uint(f);
    u += 0x7FFFu + ((u >> 16) & 1u);          // round-to-nearest-even
    return (unsigned short)(u >> 16);
}

// ------------------------- radix select (top-k threshold) -------------------

__global__ void k_zero(unsigned* __restrict__ p, int n) {
    int i = blockIdx.x * blockDim.x + threadIdx.x;
    int stride = gridDim.x * blockDim.x;
    for (; i < n; i += stride) p[i] = 0u;
}

__global__ void k_hist_hi(const float* __restrict__ s, long n, unsigned* __restrict__ hist) {
    long i = (long)blockIdx.x * blockDim.x + threadIdx.x;
    long stride = (long)gridDim.x * blockDim.x;
    for (; i < n; i += stride) {
        unsigned b = __float_as_uint(fabsf(s[i]));
        atomicAdd(&hist[b >> 16], 1u);
    }
}

__global__ void k_pick_hi(const unsigned* __restrict__ hist, unsigned long j,
                          unsigned* __restrict__ sel) {
    if (blockIdx.x == 0 && threadIdx.x == 0) {
        unsigned long cum = 0;
        for (int b = 0; b < 65536; ++b) {
            unsigned c = hist[b];
            if (cum + c > j) { sel[0] = (unsigned)b; sel[1] = (unsigned)(j - cum); return; }
            cum += c;
        }
        sel[0] = 65535u; sel[1] = 0u;
    }
}

__global__ void k_hist_lo(const float* __restrict__ s, long n,
                          const unsigned* __restrict__ sel, unsigned* __restrict__ hist) {
    unsigned hi = sel[0];
    long i = (long)blockIdx.x * blockDim.x + threadIdx.x;
    long stride = (long)gridDim.x * blockDim.x;
    for (; i < n; i += stride) {
        unsigned b = __float_as_uint(fabsf(s[i]));
        if ((b >> 16) == hi) atomicAdd(&hist[b & 0xFFFFu], 1u);
    }
}

__global__ void k_pick_lo(const unsigned* __restrict__ hist, const unsigned* __restrict__ sel,
                          unsigned* __restrict__ th) {
    if (blockIdx.x == 0 && threadIdx.x == 0) {
        unsigned long j = sel[1];
        unsigned hi = sel[0];
        unsigned long cum = 0;
        for (int b = 0; b < 65536; ++b) {
            unsigned c = hist[b];
            if (cum + c > j) { th[0] = (hi << 16) | (unsigned)b; return; }
            cum += c;
        }
        th[0] = (hi << 16) | 0xFFFFu;
    }
}

// ------------------------- fragment pack kernels ----------------------------
// xs  : [1024 row-blocks][25 ktiles][32 lanes][16 bf16]   (A-frag layout, 16x32)
// w1s : [512 n-tiles  ][25 ktiles][32 lanes][16 bf16]     (B-frag layout, 32x16)
// w2s : [256 k-pairs  ][32 lanes][16 bf16]                (B-frag layout, 32x16)

__global__ void k_pack_x(const float* __restrict__ x, uint4* __restrict__ xs) {
    int gid = blockIdx.x * blockDim.x + threadIdx.x;   // 1024*25*32 = 819200 threads
    if (gid >= 1024 * KT_N * 32) return;
    int lane = gid & 31;
    int kt   = (gid >> 5) % KT_N;
    int mb   = gid / (KT_N * 32);
    int r    = mb * 16 + (lane & 15);
    int kh   = lane >> 4;
    union { unsigned short s[16]; uint4 q[2]; } u;
    #pragma unroll
    for (int j = 0; j < 8; ++j) {
        int k0 = kt * 32 + kh * 8 + j;
        int k1 = kt * 32 + 16 + kh * 8 + j;
        u.s[j]     = (k0 < IN_DIM) ? f2bf(x[(size_t)r * IN_DIM + k0]) : (unsigned short)0;
        u.s[8 + j] = (k1 < IN_DIM) ? f2bf(x[(size_t)r * IN_DIM + k1]) : (unsigned short)0;
    }
    xs[(size_t)gid * 2 + 0] = u.q[0];
    xs[(size_t)gid * 2 + 1] = u.q[1];
}

__global__ void k_pack_w1(const float* __restrict__ w1, const float* __restrict__ s1,
                          const unsigned* __restrict__ th, uint4* __restrict__ w1s) {
    int gid = blockIdx.x * blockDim.x + threadIdx.x;   // 512*25*32 = 409600 threads
    if (gid >= 512 * KT_N * 32) return;
    unsigned T = th[0];
    int lane = gid & 31;
    int kt   = (gid >> 5) % KT_N;
    int nb   = gid / (KT_N * 32);
    int nrow = nb * 16 + (lane & 15);
    int kb   = kt * 32 + 8 * (lane >> 4);
    union { unsigned short s[16]; uint4 q[2]; } u;
    #pragma unroll
    for (int j = 0; j < 16; ++j) {
        int k = kb + (j & 7) + ((j & 8) << 1);
        unsigned short e = 0;
        if (k < IN_DIM) {
            size_t idx = (size_t)nrow * IN_DIM + k;
            unsigned bits = __float_as_uint(fabsf(s1[idx]));
            if (bits >= T) e = f2bf(w1[idx]);
        }
        u.s[j] = e;
    }
    w1s[(size_t)gid * 2 + 0] = u.q[0];
    w1s[(size_t)gid * 2 + 1] = u.q[1];
}

__global__ void k_pack_w2(const float* __restrict__ w2, const float* __restrict__ s2,
                          const unsigned* __restrict__ th, uint4* __restrict__ w2s) {
    int gid = blockIdx.x * blockDim.x + threadIdx.x;   // 256*32 = 8192 threads
    if (gid >= 256 * 32) return;
    unsigned T = th[0];
    int lane = gid & 31;
    int p    = gid >> 5;
    int c    = lane & 15;
    int kb   = 32 * p + 8 * (lane >> 4);
    union { unsigned short s[16]; uint4 q[2]; } u;
    #pragma unroll
    for (int j = 0; j < 16; ++j) {
        unsigned short e = 0;
        if (c < OUT_DIM) {
            size_t idx = (size_t)c * HIDDEN + kb + (j & 7) + ((j & 8) << 1);
            unsigned bits = __float_as_uint(fabsf(s2[idx]));
            if (bits >= T) e = f2bf(w2[idx]);
        }
        u.s[j] = e;
    }
    w2s[(size_t)gid * 2 + 0] = u.q[0];
    w2s[(size_t)gid * 2 + 1] = u.q[1];
}

// ------------------------- fused GEMM1+relu+GEMM2+log_softmax --------------
// 256 blocks x 256 threads (8 waves). Each block: 64 batch rows.
// Wave mapping: wm2 = wave&1 -> M-tiles {2*wm2, 2*wm2+1};
//               wq  = wave>>1 -> hidden quarter (2048 hidden units, 64 pairs).
// Each kt step: 2 A-frags (LDS) x 2 B-frags (global L2) -> 4 WMMAs.
// LDS: [0,102400)        x fragments (4 mtiles * 25 kt * 1KB), async-loaded
//      [102400,118784)   h staging, 2KB per wave
//      [118784,135168)   logits partials [4 quarters][4 mtiles][16][16] f32

__global__ __launch_bounds__(256, 1) void k_fused(
    const uint4* __restrict__ xs, const uint4* __restrict__ w1s,
    const uint4* __restrict__ w2s, float* __restrict__ out)
{
    extern __shared__ char smem[];
    uint4*          xtile  = (uint4*)smem;                          // 6400 uint4
    unsigned short* hstage = (unsigned short*)(smem + 102400);      // 8 waves * 1024
    float*          lbuf   = (float*)(smem + 118784);               // 4096 f32

    const int tid  = threadIdx.x;
    const int lane = tid & 31;
    const int wave = tid >> 5;
    const int wm2  = wave & 1;     // M pair: mtiles 2*wm2, 2*wm2+1
    const int wq   = wave >> 1;    // hidden quarter 0..3
    const int l15  = lane & 15;
    const int lh   = lane >> 4;

    // async global->LDS load of this block's swizzled x tile (100KB)
    {
        const uint4* src = xs + (size_t)blockIdx.x * 6400;
        #pragma unroll
        for (int i = 0; i < 25; ++i) {
            int idx = i * 256 + tid;
            __builtin_amdgcn_global_load_async_to_lds_b128(
                (AS1 v4i*)(src + idx), (AS3 v4i*)(xtile + idx), 0, 0);
        }
        asm volatile("s_wait_asynccnt 0" ::: "memory");
    }
    __syncthreads();

    unsigned short* hw = hstage + wave * 1024;  // two 16x32 bf16 tiles
    v8f lacc0{}, lacc1{};                        // logits partials per M-tile

    for (int p = 0; p < 64; ++p) {
        const int pg = wq * 64 + p;              // global hidden pair 0..255
        v8f acc00{}, acc01{}, acc10{}, acc11{};  // [mtile][ntile]
        const uint4* a0p = xtile + ((size_t)(2 * wm2) * KT_N * 32 + lane) * 2;
        const uint4* a1p = a0p + (size_t)KT_N * 32 * 2;
        const uint4* b0p = w1s + ((size_t)(pg * 2) * KT_N * 32 + lane) * 2;
        const uint4* b1p = b0p + (size_t)KT_N * 32 * 2;
        #pragma unroll 5
        for (int kt = 0; kt < KT_N; ++kt) {
            FragU a0, a1, b0, b1;
            a0.q[0] = a0p[0]; a0.q[1] = a0p[1];
            a1.q[0] = a1p[0]; a1.q[1] = a1p[1];
            b0.q[0] = b0p[0]; b0.q[1] = b0p[1];
            b1.q[0] = b1p[0]; b1.q[1] = b1p[1];
            acc00 = __builtin_amdgcn_wmma_f32_16x16x32_bf16(false, a0.v, false, b0.v,
                                                            (short)0, acc00, false, false);
            acc01 = __builtin_amdgcn_wmma_f32_16x16x32_bf16(false, a0.v, false, b1.v,
                                                            (short)0, acc01, false, false);
            acc10 = __builtin_amdgcn_wmma_f32_16x16x32_bf16(false, a1.v, false, b0.v,
                                                            (short)0, acc10, false, false);
            acc11 = __builtin_amdgcn_wmma_f32_16x16x32_bf16(false, a1.v, false, b1.v,
                                                            (short)0, acc11, false, false);
            a0p += 64; a1p += 64; b0p += 64; b1p += 64;
        }
        // relu -> bf16, stage two H(16x32) tiles row-major into wave-private LDS
        #pragma unroll
        for (int g = 0; g < 8; ++g) {
            int M = g + 8 * lh;
            float h00 = acc00[g] > 0.f ? acc00[g] : 0.f;
            float h01 = acc01[g] > 0.f ? acc01[g] : 0.f;
            float h10 = acc10[g] > 0.f ? acc10[g] : 0.f;
            float h11 = acc11[g] > 0.f ? acc11[g] : 0.f;
            hw[M * 32 + l15]            = f2bf(h00);
            hw[M * 32 + 16 + l15]       = f2bf(h01);
            hw[512 + M * 32 + l15]      = f2bf(h10);
            hw[512 + M * 32 + 16 + l15] = f2bf(h11);
        }
        asm volatile("s_wait_dscnt 0" ::: "memory");   // wave-local LDS exchange
        FragU wb, h0, h1;
        wb.q[0] = w2s[((size_t)pg * 32 + lane) * 2 + 0];
        wb.q[1] = w2s[((size_t)pg * 32 + lane) * 2 + 1];
        h0.q[0] = *(const uint4*)(hw + (l15 * 32 + lh * 8));
        h0.q[1] = *(const uint4*)(hw + (l15 * 32 + 16 + lh * 8));
        h1.q[0] = *(const uint4*)(hw + 512 + (l15 * 32 + lh * 8));
        h1.q[1] = *(const uint4*)(hw + 512 + (l15 * 32 + 16 + lh * 8));
        lacc0 = __builtin_amdgcn_wmma_f32_16x16x32_bf16(false, h0.v, false, wb.v,
                                                        (short)0, lacc0, false, false);
        lacc1 = __builtin_amdgcn_wmma_f32_16x16x32_bf16(false, h1.v, false, wb.v,
                                                        (short)0, lacc1, false, false);
    }

    // write logits partials: lbuf[((wq*4 + mt)*16 + M)*16 + c]
    #pragma unroll
    for (int g = 0; g < 8; ++g) {
        int M = g + 8 * lh;
        lbuf[((wq * 4 + 2 * wm2 + 0) * 16 + M) * 16 + l15] = lacc0[g];
        lbuf[((wq * 4 + 2 * wm2 + 1) * 16 + M) * 16 + l15] = lacc1[g];
    }
    __syncthreads();

    // 4-way combine + log_softmax over 10 classes
    if (tid < 64) {
        float v[OUT_DIM];
        float mx = -3.402823466e38f;
        #pragma unroll
        for (int c = 0; c < OUT_DIM; ++c) {
            float t = 0.f;
            #pragma unroll
            for (int q = 0; q < 4; ++q)
                t += lbuf[((q * 4) * 16 + tid) * 16 + c];   // (q*4+mt)*16+M == q*64 + tid
            v[c] = t;
            mx = t > mx ? t : mx;
        }
        float s = 0.f;
        #pragma unroll
        for (int c = 0; c < OUT_DIM; ++c) s += expf(v[c] - mx);
        float lse = mx + logf(s);
        size_t row = (size_t)blockIdx.x * 64 + tid;
        #pragma unroll
        for (int c = 0; c < OUT_DIM; ++c) out[row * OUT_DIM + c] = v[c] - lse;
    }
}

// ------------------------- launch ------------------------------------------

extern "C" void kernel_launch(void* const* d_in, const int* in_sizes, int n_in,
                              void* d_out, int out_size, void* d_ws, size_t ws_size,
                              hipStream_t stream) {
    const float* x  = (const float*)d_in[0];
    const float* w1 = (const float*)d_in[1];
    const float* s1 = (const float*)d_in[2];
    const float* w2 = (const float*)d_in[3];
    const float* s2 = (const float*)d_in[4];
    float* out = (float*)d_out;

    unsigned char* ws = (unsigned char*)d_ws;
    unsigned* hist = (unsigned*)ws;                       // 65536 u32
    unsigned* sel  = (unsigned*)(ws + 65536u * 4u);       // [0]=hi bin, [1]=local rank
    unsigned* th1  = sel + 2;
    unsigned* th2  = sel + 3;
    const size_t base = (1u << 18) + 1024u;               // 263168, aligned
    uint4* xs  = (uint4*)(ws + base);                                   // 26,214,400 B
    uint4* w1s = (uint4*)(ws + base + 26214400u);                       // 13,107,200 B
    uint4* w2s = (uint4*)(ws + base + 26214400u + 13107200u);           //    262,144 B

    const long n1 = (long)HIDDEN * IN_DIM;   // 6,422,528
    const long n2 = (long)OUT_DIM * HIDDEN;  //    81,920
    const unsigned long j1 = (unsigned long)(n1 / 2);
    const unsigned long j2 = (unsigned long)(n2 / 2);

    // threshold for scores1
    k_zero   <<<256, 256, 0, stream>>>(hist, 65536);
    k_hist_hi<<<2048, 256, 0, stream>>>(s1, n1, hist);
    k_pick_hi<<<1, 1, 0, stream>>>(hist, j1, sel);
    k_zero   <<<256, 256, 0, stream>>>(hist, 65536);
    k_hist_lo<<<2048, 256, 0, stream>>>(s1, n1, sel, hist);
    k_pick_lo<<<1, 1, 0, stream>>>(hist, sel, th1);

    // threshold for scores2
    k_zero   <<<256, 256, 0, stream>>>(hist, 65536);
    k_hist_hi<<<320, 256, 0, stream>>>(s2, n2, hist);
    k_pick_hi<<<1, 1, 0, stream>>>(hist, j2, sel);
    k_zero   <<<256, 256, 0, stream>>>(hist, 65536);
    k_hist_lo<<<320, 256, 0, stream>>>(s2, n2, sel, hist);
    k_pick_lo<<<1, 1, 0, stream>>>(hist, sel, th2);

    // bf16 fragment packing
    k_pack_x <<<3200, 256, 0, stream>>>(x, xs);
    k_pack_w1<<<1600, 256, 0, stream>>>(w1, s1, th1, w1s);
    k_pack_w2<<<32,   256, 0, stream>>>(w2, s2, th2, w2s);

    // fused MLP
    k_fused<<<256, 256, 135168, stream>>>(xs, w1s, w2s, out);
}
